// Decoder_75883482185932
// MI455X (gfx1250) — matrix operand order repaired and checked by reading, not compiled, not run on hardware
//
#include <hip/hip_runtime.h>
#include <hip/hip_bf16.h>

// ---------------------------------------------------------------------------
// Pointer-network decoder for MI455X (gfx1250), wave32 + WMMA.
//
// Key observations vs the reference:
//  * scan carry is only `ll` (a sum) -> all T=64 steps are independent.
//  * log_p[argmax] == -logsumexp(x - max)  (argmax never needed explicitly).
//  * kh/vh fold:  E@W_K1^T@Wk_m^T = E@(Wk_m*W_K1)^T  (same for vh, and
//    query = mha_out @ (W_Q*Wo_m)^T).
//
// Plan:
//  Phase A (tiny kernels): build f16 combined weights + g_ctx in workspace.
//  Phase B (main kernel): 1 workgroup per batch element b.
//    - stage node_emb[b] to LDS (f16), WMMA-compute KH/VH/K2 tiles into LDS
//      (3 x 64KB f16 -- only possible because CDNA5 has 320KB LDS per WGP).
//    - loop 4 tiles of 16 decode steps; 16 steps form the WMMA M dimension.
//      context->qh->scores->softmax->attn@VH->query->logits->(-logsumexp)
//      all with v_wmma_f32_16x16x32_f16; per-head softmax in registers with
//      16-lane shfl_xor reductions (1 head per wave, 8 waves).
// ---------------------------------------------------------------------------

#define B_  2048
#define N_  128
#define D_  256
#define H_  8
#define T_  64
#define HD_ 32

typedef __attribute__((ext_vector_type(16))) _Float16 v16h;
typedef __attribute__((ext_vector_type(8)))  float    v8f;

__device__ __forceinline__ v8f wmma32(v16h a, v16h b, v8f c) {
  // D = A(16x32) * B(32x16) + C, f32 accumulate
  return __builtin_amdgcn_wmma_f32_16x16x32_f16(
      /*neg_a=*/false, a, /*neg_b=*/false, b,
      /*c_mod=*/(short)0, c, /*reuse_a=*/false, /*reuse_b=*/false);
}

__device__ __forceinline__ v8f v8f_zero() {
  v8f c = {0.f, 0.f, 0.f, 0.f, 0.f, 0.f, 0.f, 0.f};
  return c;
}

// ---- fragment loaders (ISA 7.12.2 layouts) --------------------------------
// A (16-bit, 16x32, row-major source with leading dim ld):
//   lane 0..15 : row M=lane,    regs hold K = kc+{0..7, 16..23}
//   lane 16..31: row M=lane-16, regs hold K = kc+{8..15, 24..31}
__device__ __forceinline__ v16h load_A(const _Float16* p, int ld, int mr, int kc) {
  const int lane = threadIdx.x & 31;
  const int row  = mr + (lane & 15);
  const int kb   = kc + ((lane >> 4) << 3);
  v16h a;
#pragma unroll
  for (int j = 0; j < 8; ++j) {
    a[j]     = p[row * ld + kb + j];
    a[8 + j] = p[row * ld + kb + 16 + j];
  }
  return a;
}

// A with per-row gather from f32 global (context stage: rows = target nodes)
__device__ __forceinline__ v16h load_A_gather(const float* ebase, const int* rows, int kc) {
  const int lane = threadIdx.x & 31;
  const float* p = ebase + (size_t)rows[lane & 15] * D_;
  const int kb   = kc + ((lane >> 4) << 3);
  v16h a;
#pragma unroll
  for (int j = 0; j < 8; ++j) {
    a[j]     = (_Float16)p[kb + j];
    a[8 + j] = (_Float16)p[kb + 16 + j];
  }
  return a;
}

// B (16-bit, 32x16) where B(k,n) = W[nc+n][kc+k]  (i.e. multiply by W^T):
//   lane 0..15 : col N=lane,    K = kc+{0..7, 16..23} ; lane 16..31: K = kc+{8..15, 24..31}
__device__ __forceinline__ v16h load_B_wt(const _Float16* p, int ld, int kc, int nc) {
  const int lane = threadIdx.x & 31;
  const int n    = nc + (lane & 15);
  const int kb   = kc + ((lane >> 4) << 3);
  v16h b;
#pragma unroll
  for (int j = 0; j < 8; ++j) {
    b[j]     = p[n * ld + kb + j];
    b[8 + j] = p[n * ld + kb + 16 + j];
  }
  return b;
}

// B where B(k,n) = M[kc+k][nc+n]  (row-major, K along rows)
__device__ __forceinline__ v16h load_B_rm(const _Float16* p, int ld, int kc, int nc) {
  const int lane = threadIdx.x & 31;
  const int n    = nc + (lane & 15);
  const int kb   = kc + ((lane >> 4) << 3);
  v16h b;
#pragma unroll
  for (int j = 0; j < 8; ++j) {
    b[j]     = p[(kb + j) * ld + n];
    b[8 + j] = p[(kb + 16 + j) * ld + n];
  }
  return b;
}

// C/D f32 16x16: lane 0..15 -> rows 0..7 col=lane ; lane 16..31 -> rows 8..15
__device__ __forceinline__ void store_C_f16(_Float16* p, int ld, int mr, int nc, v8f c) {
  const int lane = threadIdx.x & 31;
  const int col  = nc + (lane & 15);
  const int rb   = mr + ((lane >> 4) << 3);
#pragma unroll
  for (int j = 0; j < 8; ++j) p[(rb + j) * ld + col] = (_Float16)c[j];
}

// ---------------------------------------------------------------------------
// Phase A kernels: f16 weight prep in workspace
// ---------------------------------------------------------------------------
__global__ void prep_convert3(const float* __restrict__ Wt,
                              const float* __restrict__ Wq,
                              const float* __restrict__ Wk2,
                              _Float16* __restrict__ hWt,
                              _Float16* __restrict__ hWq,
                              _Float16* __restrict__ hWk2) {
  int i = blockIdx.x * blockDim.x + threadIdx.x;
  if (i < D_ * D_) {
    hWt[i]  = (_Float16)Wt[i];
    hWq[i]  = (_Float16)Wq[i];
    hWk2[i] = (_Float16)Wk2[i];
  }
}

// out = A @ B, all [256,256] row-major; one thread per output element
__global__ void prep_mm_dd(const float* __restrict__ A,
                           const float* __restrict__ Bm,
                           _Float16* __restrict__ out) {
  int i = blockIdx.x * blockDim.x + threadIdx.x;
  if (i >= D_ * D_) return;
  int r = i >> 8, c = i & 255;
  float acc = 0.f;
#pragma unroll 8
  for (int k = 0; k < D_; ++k) acc += A[r * D_ + k] * Bm[k * D_ + c];
  out[i] = (_Float16)acc;
}

// hG[b][d] = sum_k graph_emb[b][k] * W_global[d][k]
__global__ void prep_gctx(const float* __restrict__ g,
                          const float* __restrict__ Wg,
                          _Float16* __restrict__ hG) {
  int i = blockIdx.x * blockDim.x + threadIdx.x;
  if (i >= B_ * D_) return;
  int b = i >> 8, d = i & 255;
  float acc = 0.f;
#pragma unroll 8
  for (int k = 0; k < D_; ++k) acc += g[b * D_ + k] * Wg[d * D_ + k];
  hG[i] = (_Float16)acc;
}

// ---------------------------------------------------------------------------
// Phase B: persistent per-batch-element decoder
// LDS map (halfs from base):
//   sKH [0, 32768)  sVH [32768, 65536)  sK2 [65536, 98304)
//   sE  [98304, 131072)   -- aliased after precompute:
//     sCtx +0 (4096) | sQ +4096 (4096) | sAttn +8192 (16384)
//     sMha +24576 (4096)  | sLogits(f32) +28672 (4096 halfs = 2048 f32)
//   misc at byte 262144: sTgt int[16], sLL float
// ---------------------------------------------------------------------------
#define LDS_BYTES 262272

extern "C" __global__ void __launch_bounds__(256)
decoder_main(const float* __restrict__ node_emb,
             const _Float16* __restrict__ hG,
             const _Float16* __restrict__ hWt,
             const _Float16* __restrict__ hWq,
             const _Float16* __restrict__ hWkc,
             const _Float16* __restrict__ hWvc,
             const _Float16* __restrict__ hWk2,
             const _Float16* __restrict__ hWqo,
             const int* __restrict__ targets,
             const int* __restrict__ masks,
             float* __restrict__ out) {
  extern __shared__ char smem[];
  _Float16* sKH = (_Float16*)smem;
  _Float16* sVH = sKH + N_ * D_;
  _Float16* sK2 = sVH + N_ * D_;
  _Float16* sE  = sK2 + N_ * D_;
  // aliases inside sE region (valid after precompute barrier)
  _Float16* sCtx    = sE;                 // 16 x 256, also reused as sQry
  _Float16* sQ      = sCtx + 16 * D_;     // 16 x 256
  _Float16* sAttn   = sQ + 16 * D_;       // 8 heads x 16 x 128
  _Float16* sMha    = sAttn + H_ * 16 * N_;      // 16 x 256
  float*    sLogits = (float*)(sMha + 16 * D_);  // 16 x 128
  int*   sTgt = (int*)(smem + 262144);
  float* sLL  = (float*)(smem + 262144 + 64);

  const int b    = blockIdx.x;
  const int tid  = threadIdx.x;
  const int wave = tid >> 5;
  const int lane = tid & 31;
  const float* eb = node_emb + (size_t)b * N_ * D_;

  if (tid == 0) *sLL = 0.f;

  // ---- stage node_emb[b] as f16 into LDS ----
  for (int i = tid; i < N_ * D_; i += 256) {
    sE[i] = (_Float16)eb[i];
  }
  __builtin_prefetch(hWkc, 0, 3);
  __syncthreads();

  // ---- precompute KH = E@Wkc^T, VH = E@Wvc^T, K2 = E@Wk2^T into LDS ----
  // 3 matrices x 8 m-tiles x 16 n-tiles = 384 tile jobs across 8 waves
  for (int job = wave; job < 384; job += 8) {
    const int mat = job >> 7;
    const int rem = job & 127;
    const int mt  = rem >> 4;
    const int nt  = rem & 15;
    const _Float16* W = (mat == 0) ? hWkc : (mat == 1) ? hWvc : hWk2;
    _Float16* dst     = (mat == 0) ? sKH  : (mat == 1) ? sVH  : sK2;
    v8f c = v8f_zero();
#pragma unroll
    for (int kc = 0; kc < D_; kc += 32)
      c = wmma32(load_A(sE, D_, mt * 16, kc), load_B_wt(W, D_, kc, nt * 16), c);
    store_C_f16(dst, D_, mt * 16, nt * 16, c);
  }
  __syncthreads();   // sE region now dead -> aliases live

  const float inv_sqrt_hd = 0.17677669529663689f;  // 1/sqrt(32)
  const float inv_sqrt_d  = 0.0625f;               // 1/sqrt(256)

  // ---- 4 tiles of 16 independent decode steps ----
  for (int t0 = 0; t0 < T_; t0 += 16) {
    if (tid < 16) sTgt[tid] = targets[(t0 + tid) * B_ + b];
    __syncthreads();

    // S0: context = tgt_emb @ W_target^T + g_ctx  -> sCtx [16 x 256]
    for (int nt = wave; nt < 16; nt += 8) {
      v8f c = v8f_zero();
#pragma unroll
      for (int kc = 0; kc < D_; kc += 32)
        c = wmma32(load_A_gather(eb, sTgt, kc), load_B_wt(hWt, D_, kc, nt * 16), c);
      const float g = (float)hG[b * D_ + nt * 16 + (lane & 15)];
#pragma unroll
      for (int j = 0; j < 8; ++j) c[j] += g;
      store_C_f16(sCtx, D_, 0, nt * 16, c);
    }
    __syncthreads();

    // S1: qh = context @ Wq_m^T -> sQ [16 x 256]
    for (int nt = wave; nt < 16; nt += 8) {
      v8f c = v8f_zero();
#pragma unroll
      for (int kc = 0; kc < D_; kc += 32)
        c = wmma32(load_A(sCtx, D_, 0, kc), load_B_wt(hWq, D_, kc, nt * 16), c);
      store_C_f16(sQ, D_, 0, nt * 16, c);
    }
    __syncthreads();

    // S2: per-head glimpse attention; wave w owns head h=w
    {
      const int h = wave;
      _Float16* sAttnH = sAttn + h * 16 * N_;
      // scores [16t x 128n], K=32 (one wmma per n-tile); A reused across tiles
      const v16h aq = load_A(sQ, D_, 0, h * HD_);
      v8f sc[8];
#pragma unroll
      for (int nt = 0; nt < 8; ++nt)
        sc[nt] = wmma32(aq, load_B_wt(sKH, D_, h * HD_, nt * 16), v8f_zero());

      // row softmax: row t = j + 8*(lane>>4); n spread over 16 lanes x 8 tiles
#pragma unroll
      for (int j = 0; j < 8; ++j) {
        float m = -3.0e38f;
#pragma unroll
        for (int nt = 0; nt < 8; ++nt) {
          sc[nt][j] *= inv_sqrt_hd;
          m = fmaxf(m, sc[nt][j]);
        }
#pragma unroll
        for (int mk = 1; mk <= 8; mk <<= 1)
          m = fmaxf(m, __shfl_xor(m, mk, 32));   // 16-lane half-wave reduce
        float s = 0.f;
#pragma unroll
        for (int nt = 0; nt < 8; ++nt) {
          float e = __expf(sc[nt][j] - m);
          sc[nt][j] = e;
          s += e;
        }
#pragma unroll
        for (int mk = 1; mk <= 8; mk <<= 1) s += __shfl_xor(s, mk, 32);
        const float inv = 1.f / s;
        const int t = j + ((lane >> 4) << 3);
#pragma unroll
        for (int nt = 0; nt < 8; ++nt)
          sAttnH[t * N_ + nt * 16 + (lane & 15)] = (_Float16)(sc[nt][j] * inv);
      }

      // out_h = attn @ VH_h  (M=16t, N=32d, K=128n) -> sMha cols [h*32, h*32+32)
#pragma unroll
      for (int dt = 0; dt < 2; ++dt) {
        v8f c = v8f_zero();
#pragma unroll
        for (int kc = 0; kc < 4; ++kc)
          c = wmma32(load_A(sAttnH, N_, 0, kc * 32),
                     load_B_rm(sVH, D_, kc * 32, h * HD_ + dt * 16), c);
        store_C_f16(sMha, D_, 0, h * HD_ + dt * 16, c);
      }
    }
    __syncthreads();

    // S3: query = mha_out @ (W_Q*Wo_m)^T -> reuse sCtx as sQry
    for (int nt = wave; nt < 16; nt += 8) {
      v8f c = v8f_zero();
#pragma unroll
      for (int kc = 0; kc < D_; kc += 32)
        c = wmma32(load_A(sMha, D_, 0, kc), load_B_wt(hWqo, D_, kc, nt * 16), c);
      store_C_f16(sCtx, D_, 0, nt * 16, c);
    }
    __syncthreads();

    // S4: logits = 10*tanh((query . key2)/sqrt(D)) - mask*1e9 -> sLogits
    {
      const int nt = wave;  // 8 n-tiles, 8 waves
      v8f c = v8f_zero();
#pragma unroll
      for (int kc = 0; kc < D_; kc += 32)
        c = wmma32(load_A(sCtx, D_, 0, kc), load_B_wt(sK2, D_, kc, nt * 16), c);
      const int col = nt * 16 + (lane & 15);
#pragma unroll
      for (int j = 0; j < 8; ++j) {
        const int t = j + ((lane >> 4) << 3);
        float x = 10.f * tanhf(c[j] * inv_sqrt_d);
        const int mk = masks[(size_t)(t0 + t) * B_ * N_ + (size_t)b * N_ + col];
        x -= (float)mk * 1.0e9f;
        sLogits[t * N_ + col] = x;
      }
    }
    __syncthreads();

    // S5: ll += log_p[argmax] == -logsumexp(x - max); one wave per 2 rows
    for (int row = wave; row < 16; row += 8) {
      float v0 = sLogits[row * N_ + lane * 4 + 0];
      float v1 = sLogits[row * N_ + lane * 4 + 1];
      float v2 = sLogits[row * N_ + lane * 4 + 2];
      float v3 = sLogits[row * N_ + lane * 4 + 3];
      float m = fmaxf(fmaxf(v0, v1), fmaxf(v2, v3));
#pragma unroll
      for (int mk = 1; mk <= 16; mk <<= 1) m = fmaxf(m, __shfl_xor(m, mk, 32));
      float s = __expf(v0 - m) + __expf(v1 - m) + __expf(v2 - m) + __expf(v3 - m);
#pragma unroll
      for (int mk = 1; mk <= 16; mk <<= 1) s += __shfl_xor(s, mk, 32);
      if (lane == 0) atomicAdd(sLL, -__logf(s));
    }
    __syncthreads();  // sLL done; buffers reusable next t-tile
  }

  if (tid == 0) out[b] = *sLL;
}

// ---------------------------------------------------------------------------
extern "C" void kernel_launch(void* const* d_in, const int* in_sizes, int n_in,
                              void* d_out, int out_size, void* d_ws, size_t ws_size,
                              hipStream_t stream) {
  const float* node_emb  = (const float*)d_in[0];
  const float* graph_emb = (const float*)d_in[1];
  const float* W_target  = (const float*)d_in[2];
  const float* W_global  = (const float*)d_in[3];
  const float* W_K1      = (const float*)d_in[4];
  const float* W_K2      = (const float*)d_in[5];
  const float* W_Q       = (const float*)d_in[6];
  const float* W_V       = (const float*)d_in[7];
  const float* Wq_m      = (const float*)d_in[8];
  const float* Wk_m      = (const float*)d_in[9];
  const float* Wv_m      = (const float*)d_in[10];
  const float* Wo_m      = (const float*)d_in[11];
  const int*   targets   = (const int*)d_in[12];
  const int*   masks     = (const int*)d_in[13];
  float* out = (float*)d_out;

  // workspace carve (f16): 6 weight mats + g_ctx
  _Float16* hWt  = (_Float16*)d_ws;
  _Float16* hWq  = hWt  + D_ * D_;
  _Float16* hWk2 = hWq  + D_ * D_;
  _Float16* hWkc = hWk2 + D_ * D_;
  _Float16* hWvc = hWkc + D_ * D_;
  _Float16* hWqo = hWvc + D_ * D_;
  _Float16* hG   = hWqo + D_ * D_;   // B_*D_ halfs

  const int thr = 256;
  prep_convert3<<<(D_ * D_ + thr - 1) / thr, thr, 0, stream>>>(
      W_target, Wq_m, W_K2, hWt, hWq, hWk2);
  prep_mm_dd<<<(D_ * D_ + thr - 1) / thr, thr, 0, stream>>>(Wk_m, W_K1, hWkc);
  prep_mm_dd<<<(D_ * D_ + thr - 1) / thr, thr, 0, stream>>>(Wv_m, W_V, hWvc);
  prep_mm_dd<<<(D_ * D_ + thr - 1) / thr, thr, 0, stream>>>(W_Q, Wo_m, hWqo);
  prep_gctx<<<(B_ * D_ + thr - 1) / thr, thr, 0, stream>>>(graph_emb, W_global, hG);

  (void)hipFuncSetAttribute((const void*)decoder_main,
                            hipFuncAttributeMaxDynamicSharedMemorySize, LDS_BYTES);
  decoder_main<<<B_, 256, LDS_BYTES, stream>>>(
      node_emb, hG, hWt, hWq, hWkc, hWvc, hWk2, hWqo, targets, masks, out);
}
